// Denoiser_9191230014134
// MI455X (gfx1250) — compile-verified, matrix operand order checked by reading, not run on hardware
//
#include <hip/hip_runtime.h>

typedef _Float16 v8h  __attribute__((ext_vector_type(8)));
typedef _Float16 v16h __attribute__((ext_vector_type(16)));
typedef float    v8f  __attribute__((ext_vector_type(8)));
typedef int      v4i  __attribute__((ext_vector_type(4)));

#define N_FFT    1024
#define HOP      256
#define CUT      513        // CUTOFF
#define BATCH    4
#define LAUD     1048576
#define PADA     512
#define NLEN     1049600    // padded audio length = LAUD + N_FFT
#define APAD_LEN 1065728    // covers frames up to 4159 (65 groups of 64) + window
#define KFP      528        // padded freq rows (33*16), valid 0..512
#define CPAD     1056       // padded channels (33*32): Re at [0,512], Im at [528,1040]
#define FROWS    4116       // rec_buf frame rows: 3 leading zeros + 4097 + slack
#define K2       4224       // GEMM2 K = 4*CPAD
#define BROW     1064       // LDS B-panel row stride in f16 (bank-conflict-free)
#define TWO_PI_OVER_N 0.006135923151542565f   // 2*pi/1024

#if defined(__HIP_DEVICE_COMPILE__) && __has_builtin(__builtin_amdgcn_global_load_async_to_lds_b128)
#define HAVE_ASYNC_LDS 1
#else
#define HAVE_ASYNC_LDS 0
#endif

#if HAVE_ASYNC_LDS
typedef __attribute__((address_space(1))) v4i* gvec_ptr;
typedef __attribute__((address_space(3))) v4i* lvec_ptr;
#endif

__device__ __forceinline__ v8f wmma16(v16h a, v16h b, v8f c) {
  return __builtin_amdgcn_wmma_f32_16x16x32_f16(false, a, false, b,
                                                (short)0, c, false, false);
}

// Assemble an A/B fragment: lane half `sel` holds K {0..7,16..23} (sel=0)
// or {8..15,24..31} (sel=1) of the 32-wide K tile starting at `base`.
__device__ __forceinline__ v16h load_frag(const _Float16* base, int sel) {
  v8h lo = *(const v8h*)(base + sel * 8);
  v8h hi = *(const v8h*)(base + 16 + sel * 8);
  return __builtin_shufflevector(lo, hi, 0,1,2,3,4,5,6,7,8,9,10,11,12,13,14,15);
}

// ---------------- prep kernels ----------------

__global__ void k_prep_apad(const float* __restrict__ audio,
                            _Float16* __restrict__ apad) {
  long idx = (long)blockIdx.x * blockDim.x + threadIdx.x;
  const long total = (long)BATCH * APAD_LEN;
  if (idx >= total) return;
  int b = (int)(idx / APAD_LEN);
  int i = (int)(idx % APAD_LEN);
  float v = 0.0f;
  if (i < NLEN) {
    int j = i - PADA;
    if (j < 0) j = -j;
    if (j >= LAUD) j = 2 * LAUD - 2 - j;
    v = audio[(long)b * LAUD + j];
  }
  apad[idx] = (_Float16)v;
}

__global__ void k_prep_fwd(_Float16* __restrict__ fwdRe,
                           _Float16* __restrict__ fwdIm) {
  int idx = blockIdx.x * blockDim.x + threadIdx.x;
  if (idx >= KFP * N_FFT) return;
  int kf = idx >> 10, n = idx & 1023;
  float win = 0.5f - 0.5f * __cosf((float)n * TWO_PI_OVER_N);
  float re = 0.0f, im = 0.0f;
  if (kf <= 512) {
    int ph = (kf * n) & 1023;            // exact argument reduction
    float a = (float)ph * TWO_PI_OVER_N;
    re =  __cosf(a) * win;
    im = -__sinf(a) * win;
  }
  fwdRe[idx] = (_Float16)re;
  fwdIm[idx] = (_Float16)im;
}

// Inverse basis, transposed for GEMM2: Bt[p][k], k = j*CPAD + c,
// value = pinv-row (c_k/N * cos / -sin) * win at n = p + 256*j.
__global__ void k_prep_bt(_Float16* __restrict__ Bt) {
  int idx = blockIdx.x * blockDim.x + threadIdx.x;
  if (idx >= 256 * K2) return;
  int p = idx / K2;
  int k = idx % K2;
  int j = k / CPAD, c = k % CPAD;
  int n = p + 256 * j;                   // 0..1023
  float win = 0.5f - 0.5f * __cosf((float)n * TWO_PI_OVER_N);
  float v = 0.0f;
  int kf = -1, isIm = 0;
  if (c < CUT) { kf = c; }
  else if (c >= 528 && c <= 1040) { kf = c - 528; isIm = 1; }
  if (kf >= 0) {
    float ck = (kf == 0 || kf == 512) ? 1.0f : 2.0f;
    int ph = (kf * n) & 1023;
    float a = (float)ph * TWO_PI_OVER_N;
    float tr = isIm ? -__sinf(a) : __cosf(a);
    v = (ck * (1.0f / 1024.0f)) * tr * win;
  }
  Bt[idx] = (_Float16)v;
}

__global__ void k_zero16(uint4* __restrict__ p, long n16) {
  long idx = (long)blockIdx.x * blockDim.x + threadIdx.x;
  if (idx < n16) p[idx] = make_uint4(0u, 0u, 0u, 0u);
}

// ---------------- GEMM1: forward STFT + spectral threshold ----------------
// One wave per 16(freq) x 64(frames) tile: A (Re/Im basis) fragments are
// reused across 4 B fragments -> 12 b128 loads per 8 WMMAs.

__global__ void __launch_bounds__(256)
k_fwd_gemm(const _Float16* __restrict__ apad,
           const _Float16* __restrict__ fwdRe,
           const _Float16* __restrict__ fwdIm,
           const float* __restrict__ bias,
           _Float16* __restrict__ rec) {
  const int lane = threadIdx.x & 31;
  const int wid  = threadIdx.x >> 5;
  const int fg   = blockIdx.x * 8 + wid;   // 64-frame group, 0..64 valid
  const int kt   = blockIdx.y;             // freq tile 0..32
  const int b    = blockIdx.z;
  if (fg >= 65) return;                    // wave-uniform; EXEC stays full
  const int f0  = fg * 64;
  const int kf0 = kt * 16;
  const int sel = lane >> 4;
  const int mr  = lane & 15;

  const _Float16* aR = fwdRe + (size_t)(kf0 + mr) * N_FFT;
  const _Float16* aI = fwdIm + (size_t)(kf0 + mr) * N_FFT;
  const _Float16* bp = apad + (size_t)b * APAD_LEN + (size_t)(f0 + mr) * HOP;

  v8f accR[4] = {{0}, {0}, {0}, {0}};
  v8f accI[4] = {{0}, {0}, {0}, {0}};
  #pragma unroll 2
  for (int k0 = 0; k0 < N_FFT; k0 += 32) {
    v16h av = load_frag(aR + k0, sel);
    v16h iv = load_frag(aI + k0, sel);
    #pragma unroll
    for (int i = 0; i < 4; ++i) {
      v16h bv = load_frag(bp + i * (16 * HOP) + k0, sel);
      accR[i] = wmma16(av, bv, accR[i]);
      accI[i] = wmma16(iv, bv, accI[i]);
    }
  }

  #pragma unroll
  for (int i = 0; i < 4; ++i) {
    const int f = f0 + i * 16 + mr;        // this lane's frame (N = lane&15)
    if (f <= 4096) {
      size_t rowb = ((size_t)b * FROWS + (size_t)(f + 3)) * CPAD;
      #pragma unroll
      for (int v = 0; v < 8; ++v) {
        int kf = kf0 + v + 8 * sel;        // M = v + 8*(lane>=16)
        if (kf <= 512) {
          float re = accR[i][v], im = accI[i][v];
          float mag = __builtin_sqrtf(re * re + im * im);
          float sub = 0.1f * bias[kf];
          float ratio = (mag > 0.0f) ? fmaxf(mag - sub, 0.0f) / mag : 0.0f;
          rec[rowb + (size_t)kf]         = (_Float16)(ratio * re);
          rec[rowb + (size_t)(528 + kf)] = (_Float16)(ratio * im);
        }
      }
    }
  }
}

// ---------------- GEMM2: inverse STFT as one GEMM + analytic ws ----------
// All 8 waves of a WG share the same 16 phases: stage each j-block of the
// B matrix (16 x 1056 f16, padded rows) into LDS via async global->LDS
// (fallback: load + ds_store), then read B fragments with ds_load_b128.

__global__ void __launch_bounds__(256)
k_inv_gemm(const _Float16* __restrict__ rec,
           const _Float16* __restrict__ Bt,
           float* __restrict__ out) {
  __shared__ _Float16 bpanel[16 * BROW];   // 34,048 bytes

  const int tid  = threadIdx.x;
  const int lane = tid & 31;
  const int wid  = tid >> 5;
  int mt = blockIdx.x * 8 + wid;           // frame-base tile
  if (mt > 256) mt = 256;                  // clamp: duplicate work, keeps waves alive
  const int nt   = blockIdx.y;             // phase tile 0..15
  const int b    = blockIdx.z;
  const int f0b  = mt * 16;
  const int sel  = lane >> 4;
  const int lr   = lane & 15;
  const int p    = nt * 16 + lr;           // phase (B column for this lane)
  const size_t rb = (size_t)b * FROWS;

  v8f acc = {0};
  for (int j = 0; j < 4; ++j) {
    // ---- cooperative stage of B panel j: rows p=nt*16..+15, cols j*1056..+1055
    __syncthreads();                       // previous j's reads complete
    for (int ch = tid; ch < 16 * 132; ch += 256) {   // 132 b128 chunks per row
      int row = ch / 132, col = ch - row * 132;
      const _Float16* gsrc = Bt + ((size_t)(nt * 16 + row)) * K2
                                + (size_t)j * CPAD + (size_t)col * 8;
      _Float16* ldst = &bpanel[row * BROW + col * 8];
#if HAVE_ASYNC_LDS
      __builtin_amdgcn_global_load_async_to_lds_b128(
          (gvec_ptr)gsrc, (lvec_ptr)ldst, 0, 0);
#else
      *(v8h*)ldst = *(const v8h*)gsrc;
#endif
    }
#if HAVE_ASYNC_LDS
#if __has_builtin(__builtin_amdgcn_s_wait_asynccnt)
    __builtin_amdgcn_s_wait_asynccnt(0);
#else
    asm volatile("s_wait_asynccnt 0" ::: "memory");
#endif
#endif
    __syncthreads();

    // ---- 33 K-tiles of WMMA against the staged panel
    const _Float16* ap = rec + (rb + (size_t)(f0b + lr + 3 - j)) * CPAD;
    const _Float16* bq = &bpanel[lr * BROW];
    #pragma unroll 3
    for (int ct = 0; ct < 33; ++ct) {
      int c0 = ct * 32;
      v16h av = load_frag(ap + c0, sel);
      v16h bv = load_frag(bq + c0, sel);   // ds_load_b128 x2, conflict-free
      acc = wmma16(av, bv, acc);
    }
  }

  #pragma unroll
  for (int v = 0; v < 8; ++v) {
    int  f0 = f0b + v + 8 * sel;           // M = row = frame base
    long t  = (long)f0 * 256 + p;
    long i  = t - PADA;                    // crop
    if (f0 <= 4099 && i >= 0 && i < LAUD) {
      float ws = 0.0f;
      #pragma unroll
      for (int j = 0; j < 4; ++j) {
        int fr = f0 - j;
        if (fr >= 0 && fr <= 4096) {
          float w = 0.5f - 0.5f * __cosf((float)(p + 256 * j) * TWO_PI_OVER_N);
          ws += w * w;
        }
      }
      float val = acc[v];
      out[(size_t)b * LAUD + i] = (ws > 1.1754944e-38f) ? val / ws : val;
    }
  }
}

// ---------------- host ----------------

extern "C" void kernel_launch(void* const* d_in, const int* in_sizes, int n_in,
                              void* d_out, int out_size, void* d_ws, size_t ws_size,
                              hipStream_t stream) {
  const float* audio = (const float*)d_in[0];   // (4, 1048576) f32
  const float* bias  = (const float*)d_in[1];   // (1, 513, 1)  f32
  float* out = (float*)d_out;                   // (4, 1048576) f32

  char* w = (char*)d_ws;
  size_t off = 0;
  auto take = [&](size_t bytes) -> char* {
    char* p = w + off;
    off = (off + bytes + 255) & ~(size_t)255;
    return p;
  };
  _Float16* apad  = (_Float16*)take((size_t)BATCH * APAD_LEN * 2);
  _Float16* fwdRe = (_Float16*)take((size_t)KFP * N_FFT * 2);
  _Float16* fwdIm = (_Float16*)take((size_t)KFP * N_FFT * 2);
  _Float16* Bt    = (_Float16*)take((size_t)256 * K2 * 2);
  _Float16* rec   = (_Float16*)take((size_t)BATCH * FROWS * CPAD * 2);

  {
    long total = (long)BATCH * APAD_LEN;
    k_prep_apad<<<dim3((unsigned)((total + 255) / 256)), dim3(256), 0, stream>>>(audio, apad);
  }
  {
    int total = KFP * N_FFT;
    k_prep_fwd<<<dim3((total + 255) / 256), dim3(256), 0, stream>>>(fwdRe, fwdIm);
  }
  {
    int total = 256 * K2;
    k_prep_bt<<<dim3((total + 255) / 256), dim3(256), 0, stream>>>(Bt);
  }
  {
    long n16 = ((long)BATCH * FROWS * CPAD * 2) / 16;
    k_zero16<<<dim3((unsigned)((n16 + 255) / 256)), dim3(256), 0, stream>>>((uint4*)rec, n16);
  }
  k_fwd_gemm<<<dim3(9, 33, BATCH), dim3(256), 0, stream>>>(apad, fwdRe, fwdIm, bias, rec);
  k_inv_gemm<<<dim3(33, 16, BATCH), dim3(256), 0, stream>>>(rec, Bt, out);
}
